// GCNCombiner_9010841387285
// MI455X (gfx1250) — compile-verified, matrix operand order checked by reading, not run on hardware
//
#include <hip/hip_runtime.h>
#include <hip/hip_bf16.h>

// ---------------------------------------------------------------------------
// GCN combiner head for MI455X (gfx1250, wave32, WMMA).
// Dominant cost: three embed MLP GEMM chains (~83 GFLOP) -> bf16 WMMA.
// Round 2: pre-transposed bf16 weights (kills scalar LDS scatter),
//          double-buffered software-pipelined GEMM main loop.
// ---------------------------------------------------------------------------

typedef __bf16 bf16_t;
typedef __attribute__((ext_vector_type(16))) __bf16 v16bf;
typedef __attribute__((ext_vector_type(8)))  __bf16 v8bf;
typedef __attribute__((ext_vector_type(2)))  __bf16 v2bf;
typedef __attribute__((ext_vector_type(8)))  float  v8f;

#define GEMM_BM 128
#define GEMM_BN 128
#define GEMM_BK 32
#define LDS_STRIDE (GEMM_BK + 8)      // bf16 elems; row pitch 80B (16B aligned)
#define LDS_TILE   (GEMM_BM * LDS_STRIDE)

// ---------------------------------------------------------------------------
// f32 -> bf16 streaming convert
// ---------------------------------------------------------------------------
__global__ void cvt_f32_bf16(const float* __restrict__ in, bf16_t* __restrict__ out, long n) {
    long i = (long)blockIdx.x * blockDim.x + threadIdx.x;
    long stride = (long)gridDim.x * blockDim.x;
    for (; i < n; i += stride) out[i] = (bf16_t)in[i];
}

// f32 [R,C] -> bf16 [C,R] transpose-convert (weights only; run once)
__global__ void cvt_transpose_bf16(const float* __restrict__ in, bf16_t* __restrict__ out,
                                   int R, int C) {
    int id = blockIdx.x * 256 + threadIdx.x;
    if (id >= R * C) return;
    int c = id / R, r = id - c * R;           // out index id = c*R + r
    out[id] = (bf16_t)in[(long)r * C + c];
}

// ---------------------------------------------------------------------------
// Tiled bf16 GEMM with v_wmma_f32_16x16x32_bf16.
//   C[m,n] = sum_k A[m,k]*Bt[n,k] + bias[n]
// A: [M,K] row-major bf16.  Bt: [N,K] row-major bf16 (pre-transposed weights).
// 256 threads = 8 waves; wave (r=wid>>1, c=wid&1) owns rows r*32..+31,
// cols c*64..+63 => 2x4 WMMA accumulators (64 f32 VGPRs/lane).
// Double-buffered LDS; next tile's global loads are issued before the WMMAs
// and drained into the alternate buffer after them (s_wait_loadcnt overlaps
// with matrix math).
// Output remap: idx = (m>>rbShift)*CS + (m&mask)*ldc + n  (scatters G2 output
// into the [B, 1536, C] token-concat layout).
// Requires M%128==0, N%128==0, K%32==0 (true for all call sites).
// ---------------------------------------------------------------------------
template<bool RELU, bool OUT_BF16>
__global__ __launch_bounds__(256)
void gemm_bf16_wmma(const bf16_t* __restrict__ A,
                    const bf16_t* __restrict__ Bt,
                    const float* __restrict__ bias,
                    void* __restrict__ Cout,
                    int M, int N, int K,
                    int rbShift, long CS, long ldc)
{
    __shared__ alignas(16) bf16_t sA[2 * LDS_TILE];
    __shared__ alignas(16) bf16_t sB[2 * LDS_TILE];   // [n][k]

    const int tid  = threadIdx.x;
    const int lane = tid & 31;
    const int wid  = tid >> 5;
    const int half = lane >> 4;   // K-half select per ISA A/B lane layout
    const int r16  = lane & 15;
    const int wm   = (wid >> 1) * 32;
    const int wn   = (wid & 1) * 64;

    const long m0 = (long)blockIdx.x * GEMM_BM;
    const long n0 = (long)blockIdx.y * GEMM_BN;

    // per-thread staging coords: 16 bf16 of A row + 16 bf16 of Bt row per tile
    const int lrow = tid >> 1;            // 0..127
    const int lkc  = (tid & 1) * 16;      // 0 or 16
    const bf16_t* gA = A  + (m0 + lrow) * (long)K + lkc;
    const bf16_t* gB = Bt + (n0 + lrow) * (long)K + lkc;
    const int ldsOff = lrow * LDS_STRIDE + lkc;

    const int nk = K / GEMM_BK;
    v8f acc[2][4] = {};
    v8bf ra0, ra1, rb0, rb1;

    // prologue: fetch + stage tile 0 into buffer 0
    ra0 = *(const v8bf*)(gA);      ra1 = *(const v8bf*)(gA + 8);
    rb0 = *(const v8bf*)(gB);      rb1 = *(const v8bf*)(gB + 8);
    *(v8bf*)(&sA[ldsOff])     = ra0;  *(v8bf*)(&sA[ldsOff + 8]) = ra1;
    *(v8bf*)(&sB[ldsOff])     = rb0;  *(v8bf*)(&sB[ldsOff + 8]) = rb1;

    for (int ks = 0; ks < nk; ++ks) {
        const int cur = ks & 1;
        __syncthreads();

        // issue next tile's global loads (no wait yet)
        const bool more = (ks + 1 < nk);
        if (more) {
            const bf16_t* pa = gA + (ks + 1) * GEMM_BK;
            const bf16_t* pb = gB + (ks + 1) * GEMM_BK;
            ra0 = *(const v8bf*)(pa);  ra1 = *(const v8bf*)(pa + 8);
            rb0 = *(const v8bf*)(pb);  rb1 = *(const v8bf*)(pb + 8);
        }
        if (ks + 2 < nk) {
            __builtin_prefetch(gA + (ks + 2) * GEMM_BK, 0, 3);
            __builtin_prefetch(gB + (ks + 2) * GEMM_BK, 0, 3);
        }

        // fragments: lanes 0-15 hold K {0..7,16..23}; lanes 16-31 {8..15,24..31}
        const bf16_t* bufA = &sA[cur * LDS_TILE];
        const bf16_t* bufB = &sB[cur * LDS_TILE];
        v16bf afrag[2];
        #pragma unroll
        for (int i = 0; i < 2; ++i) {
            const bf16_t* p = &bufA[(wm + i * 16 + r16) * LDS_STRIDE + half * 8];
            v8bf lo = *(const v8bf*)(p);
            v8bf hi = *(const v8bf*)(p + 16);
            afrag[i] = __builtin_shufflevector(lo, hi, 0,1,2,3,4,5,6,7,8,9,10,11,12,13,14,15);
        }
        v16bf bfrag[4];
        #pragma unroll
        for (int j = 0; j < 4; ++j) {
            const bf16_t* p = &bufB[(wn + j * 16 + r16) * LDS_STRIDE + half * 8];
            v8bf lo = *(const v8bf*)(p);
            v8bf hi = *(const v8bf*)(p + 16);
            bfrag[j] = __builtin_shufflevector(lo, hi, 0,1,2,3,4,5,6,7,8,9,10,11,12,13,14,15);
        }

        #pragma unroll
        for (int i = 0; i < 2; ++i)
            #pragma unroll
            for (int j = 0; j < 4; ++j)
                acc[i][j] = __builtin_amdgcn_wmma_f32_16x16x32_bf16(
                    false, afrag[i], false, bfrag[j], (short)0, acc[i][j], false, false);

        // drain next tile into the alternate buffer (wait_loadcnt lands here,
        // after the WMMAs)
        if (more) {
            bf16_t* da = &sA[(cur ^ 1) * LDS_TILE + ldsOff];
            bf16_t* db = &sB[(cur ^ 1) * LDS_TILE + ldsOff];
            *(v8bf*)(da) = ra0;  *(v8bf*)(da + 8) = ra1;
            *(v8bf*)(db) = rb0;  *(v8bf*)(db + 8) = rb1;
        }
    }

    // epilogue: C layout VGPR v -> row v + 8*half, col r16
    const int rbMask = (1 << rbShift) - 1;
    #pragma unroll
    for (int i = 0; i < 2; ++i) {
        #pragma unroll
        for (int j = 0; j < 4; ++j) {
            int n = (int)n0 + wn + j * 16 + r16;
            float bv = bias[n];
            #pragma unroll
            for (int v = 0; v < 8; ++v) {
                int m = (int)m0 + wm + i * 16 + half * 8 + v;
                float val = acc[i][j][v] + bv;
                if (RELU) val = fmaxf(val, 0.0f);
                long idx = (long)(m >> rbShift) * CS + (long)(m & rbMask) * ldc + n;
                if (OUT_BF16) ((bf16_t*)Cout)[idx] = (bf16_t)val;
                else          ((float*)Cout)[idx]  = val;
            }
        }
    }
}

// ---------------------------------------------------------------------------
// pool0: out[b,c,j] = sum_t hs[b,t,c] * p0w[t,j] + p0b[j]
// hs tokens 0..1535 from bf16 buffer, 1536..2047 directly from fp32 x3.
// Each thread streams 2 adjacent channels (4B bf16 / 8B f32 loads).
// ---------------------------------------------------------------------------
__global__ __launch_bounds__(256)
void pool0_kernel(const bf16_t* __restrict__ hs_bf, const float* __restrict__ x3,
                  const float* __restrict__ p0w, const float* __restrict__ p0b,
                  float* __restrict__ out)
{
    __shared__ float w[512 * 8];
    int c0 = (blockIdx.x * 256 + threadIdx.x) * 2;
    int b = blockIdx.y;
    float acc0[8] = {0.f,0.f,0.f,0.f,0.f,0.f,0.f,0.f};
    float acc1[8] = {0.f,0.f,0.f,0.f,0.f,0.f,0.f,0.f};

    for (int chunk = 0; chunk < 4; ++chunk) {
        __syncthreads();
        #pragma unroll
        for (int e = 0; e < 16; ++e) {
            int idx = e * 256 + threadIdx.x;
            w[idx] = p0w[chunk * 4096 + idx];
        }
        __syncthreads();

        if (chunk < 3) {
            const bf16_t* hp = hs_bf + ((long)b * 1536 + chunk * 512) * 1024 + c0;
            for (int t = 0; t < 512; ++t) {
                v2bf v = *(const v2bf*)(hp + (long)t * 1024);
                float v0 = (float)v[0], v1 = (float)v[1];
                const float* wr = &w[t * 8];
                #pragma unroll
                for (int j = 0; j < 8; ++j) {
                    acc0[j] = fmaf(v0, wr[j], acc0[j]);
                    acc1[j] = fmaf(v1, wr[j], acc1[j]);
                }
            }
        } else {
            const float* xp = x3 + (long)b * 512 * 1024 + c0;
            for (int t = 0; t < 512; ++t) {
                float2 v = *(const float2*)(xp + (long)t * 1024);
                const float* wr = &w[t * 8];
                #pragma unroll
                for (int j = 0; j < 8; ++j) {
                    acc0[j] = fmaf(v.x, wr[j], acc0[j]);
                    acc1[j] = fmaf(v.y, wr[j], acc1[j]);
                }
            }
        }
    }
    float* op = out + ((long)b * 1024 + c0) * 8;
    #pragma unroll
    for (int j = 0; j < 8; ++j) {
        op[j]     = acc0[j] + p0b[j];
        op[8 + j] = acc1[j] + p0b[j];
    }
}

// ---------------------------------------------------------------------------
// mean over d of q_w/k_w rows (+ mean of q_b/k_b into slots 2048/2049).
// Collapses mean(conv_q(hs)) to a single 1024-dot per (b,j).
// ---------------------------------------------------------------------------
__global__ void qk_mean_kernel(const float* __restrict__ qw, const float* __restrict__ qb,
                               const float* __restrict__ kw, const float* __restrict__ kb,
                               float* __restrict__ out)
{
    int id = blockIdx.x * 256 + threadIdx.x;
    if (id < 1024) {
        float s = 0.f;
        for (int d = 0; d < 256; ++d) s += qw[(long)id * 256 + d];
        out[id] = s * (1.f / 256.f);
    } else if (id < 2048) {
        int c = id - 1024;
        float s = 0.f;
        for (int d = 0; d < 256; ++d) s += kw[(long)c * 256 + d];
        out[id] = s * (1.f / 256.f);
    } else if (id == 2048) {
        float s = 0.f;
        for (int d = 0; d < 256; ++d) s += qb[d];
        out[2048] = s * (1.f / 256.f);
    } else if (id == 2049) {
        float s = 0.f;
        for (int d = 0; d < 256; ++d) s += kb[d];
        out[2049] = s * (1.f / 256.f);
    }
}

// q1k1[b*16 + r]: r<8 -> q1[b,r], r>=8 -> k1[b,r-8]
__global__ void q1k1_kernel(const float* __restrict__ pool_hs, const float* __restrict__ qkm,
                            float* __restrict__ q1k1)
{
    int id = blockIdx.x * 256 + threadIdx.x;
    if (id >= 1024) return;
    int b = id >> 4, r = id & 15, isK = r >> 3, j = r & 7;
    const float* hp = pool_hs + (long)b * 8192 + j;
    const float* mp = qkm + isK * 1024;
    float s = 0.f;
    for (int c = 0; c < 1024; ++c) s = fmaf(hp[(long)c * 8], mp[c], s);
    q1k1[id] = s + qkm[2048 + isK];
}

__global__ void a1_kernel(const float* __restrict__ q1k1, const float* __restrict__ adj1,
                          const float* __restrict__ alpha, float* __restrict__ A1)
{
    int b = blockIdx.x, t = threadIdx.x;        // 64 threads: t = i*8 + j
    int i = t >> 3, j = t & 7;
    float q = q1k1[b * 16 + i];
    float k = q1k1[b * 16 + 8 + j];
    A1[b * 64 + t] = adj1[t] + tanhf(q - k) * alpha[0];
}

// pool_hs [B,C,8] f32 -> hsT [B*8, C] bf16 (rows = b*8+j) for the conv1 GEMM
__global__ void hsT_kernel(const float* __restrict__ pool_hs, bf16_t* __restrict__ hsT)
{
    int id = blockIdx.x * 256 + threadIdx.x;    // 512*1024
    if (id >= 512 * 1024) return;
    int row = id >> 10, c = id & 1023;
    int b = row >> 3, j = row & 7;
    hsT[id] = (bf16_t)pool_hs[((long)b * 1024 + c) * 8 + j];
}

// hs3T[b*8+k, d] = sum_j hs2T[b*8+j, d] * A1[b,j,k]
__global__ __launch_bounds__(256)
void graph_kernel(const float* __restrict__ hs2T, const float* __restrict__ A1,
                  float* __restrict__ hs3T)
{
    __shared__ float a[64];
    int b = blockIdx.y;
    int d = blockIdx.x * 256 + threadIdx.x;
    if (threadIdx.x < 64) a[threadIdx.x] = A1[b * 64 + threadIdx.x];
    __syncthreads();
    float h[8];
    #pragma unroll
    for (int j = 0; j < 8; ++j) h[j] = hs2T[(long)(b * 8 + j) * 1024 + d];
    #pragma unroll
    for (int k = 0; k < 8; ++k) {
        float s = 0.f;
        #pragma unroll
        for (int j = 0; j < 8; ++j) s = fmaf(h[j], a[j * 8 + k], s);
        hs3T[(long)(b * 8 + k) * 1024 + d] = s;
    }
}

// BatchNorm (training stats over B*J=512 rows per channel) -> scale/shift
__global__ void bn_stats_kernel(const float* __restrict__ hs3T, const float* __restrict__ gamma,
                                const float* __restrict__ beta, float* __restrict__ scale,
                                float* __restrict__ shift)
{
    int d = blockIdx.x * 256 + threadIdx.x;
    if (d >= 1024) return;
    float s = 0.f, ss = 0.f;
    for (int r = 0; r < 512; ++r) {
        float v = hs3T[(long)r * 1024 + d];
        s += v;
        ss = fmaf(v, v, ss);
    }
    float mean = s * (1.f / 512.f);
    float var  = ss * (1.f / 512.f) - mean * mean;
    float sc   = gamma[d] * rsqrtf(var + 1e-5f);
    scale[d] = sc;
    shift[d] = beta[d] - mean * sc;
}

// fused BN-apply + pool1 over joints -> hs4[b,d]
__global__ void pool1_kernel(const float* __restrict__ hs3T, const float* __restrict__ scale,
                             const float* __restrict__ shift, const float* __restrict__ p1w,
                             const float* __restrict__ p1b, float* __restrict__ hs4)
{
    int id = blockIdx.x * 256 + threadIdx.x;    // 64*1024
    int b = id >> 10, d = id & 1023;
    float sc = scale[d], sh = shift[d];
    float s = 0.f;
    #pragma unroll
    for (int j = 0; j < 8; ++j) {
        float v = hs3T[(long)(b * 8 + j) * 1024 + d];
        s = fmaf(v * sc + sh, p1w[j], s);
    }
    hs4[id] = s + p1b[0];
}

// classifier: out[b,n] = sum_d hs4[b,d]*cls_w[d,n] + cls_b[n]   (N=200)
__global__ __launch_bounds__(256)
void cls_kernel(const float* __restrict__ hs4, const float* __restrict__ cw,
                const float* __restrict__ cb, float* __restrict__ out)
{
    __shared__ float h[1024];
    int b = blockIdx.x;
    for (int e = threadIdx.x; e < 1024; e += 256) h[e] = hs4[b * 1024 + e];
    __syncthreads();
    int n = threadIdx.x;
    if (n < 200) {
        float s = 0.f;
        for (int d = 0; d < 1024; ++d) s = fmaf(h[d], cw[(long)d * 200 + n], s);
        out[b * 200 + n] = s + cb[n];
    }
}

// ---------------------------------------------------------------------------
extern "C" void kernel_launch(void* const* d_in, const int* in_sizes, int n_in,
                              void* d_out, int out_size, void* d_ws, size_t ws_size,
                              hipStream_t stream)
{
    const float* x0 = (const float*)d_in[0];
    const float* x1 = (const float*)d_in[1];
    const float* x2 = (const float*)d_in[2];
    const float* x3 = (const float*)d_in[3];
    const float* e_w1[3] = {(const float*)d_in[4], (const float*)d_in[8],  (const float*)d_in[12]};
    const float* e_b1[3] = {(const float*)d_in[5], (const float*)d_in[9],  (const float*)d_in[13]};
    const float* e_w2[3] = {(const float*)d_in[6], (const float*)d_in[10], (const float*)d_in[14]};
    const float* e_b2[3] = {(const float*)d_in[7], (const float*)d_in[11], (const float*)d_in[15]};
    const float* pool0_w  = (const float*)d_in[16];
    const float* pool0_b  = (const float*)d_in[17];
    const float* adj1     = (const float*)d_in[18];
    const float* conv1_w  = (const float*)d_in[19];
    const float* conv1_b  = (const float*)d_in[20];
    const float* q_w      = (const float*)d_in[21];
    const float* q_b      = (const float*)d_in[22];
    const float* k_w      = (const float*)d_in[23];
    const float* k_b      = (const float*)d_in[24];
    const float* alpha1   = (const float*)d_in[25];
    const float* bn_gamma = (const float*)d_in[26];
    const float* bn_beta  = (const float*)d_in[27];
    const float* pool1_w  = (const float*)d_in[28];
    const float* pool1_b  = (const float*)d_in[29];
    const float* cls_w    = (const float*)d_in[30];
    const float* cls_b    = (const float*)d_in[31];
    float* out = (float*)d_out;

    // ---- workspace carve (~305 MB total) ----
    char* wsb = (char*)d_ws;
    size_t off = 0;
    auto carve = [&](size_t bytes) -> char* {
        char* p = wsb + off;
        off += (bytes + 255) & ~(size_t)255;
        return p;
    };
    const long nx[3]  = {64L * 512 * 128, 64L * 512 * 256, 64L * 512 * 512};
    const long nw1[3] = {128L * 128, 256L * 256, 512L * 512};
    const long nw2[3] = {128L * 1024, 256L * 1024, 512L * 1024};

    bf16_t* xb[3];   for (int s = 0; s < 3; ++s) xb[s]   = (bf16_t*)carve((size_t)nx[s] * 2);
    bf16_t* wb1t[3]; for (int s = 0; s < 3; ++s) wb1t[s] = (bf16_t*)carve((size_t)nw1[s] * 2);
    bf16_t* wb2t[3]; for (int s = 0; s < 3; ++s) wb2t[s] = (bf16_t*)carve((size_t)nw2[s] * 2);
    bf16_t* conv1wbt = (bf16_t*)carve(1024ull * 1024 * 2);
    bf16_t* ybf      = (bf16_t*)carve(32768ull * 512 * 2);         // relu intermediate (max stage)
    bf16_t* hs_bf    = (bf16_t*)carve(64ull * 1536 * 1024 * 2);    // embedded tokens (192 MB)
    float*  pool_hs  = (float*)carve(64ull * 1024 * 8 * 4);        // [B,C,8]
    float*  qkm      = (float*)carve(2052ull * 4);
    float*  q1k1     = (float*)carve(1024ull * 4);
    float*  A1b      = (float*)carve(4096ull * 4);
    bf16_t* hsT      = (bf16_t*)carve(512ull * 1024 * 2);
    float*  hs2T     = (float*)carve(512ull * 1024 * 4);
    float*  hs3T     = (float*)carve(512ull * 1024 * 4);
    float*  bnscale  = (float*)carve(1024ull * 4);
    float*  bnshift  = (float*)carve(1024ull * 4);
    float*  hs4      = (float*)carve(64ull * 1024 * 4);

    auto cvt = [&](const float* in, bf16_t* o, long n) {
        long blocks = (n + 255) / 256;
        if (blocks > 8192) blocks = 8192;
        cvt_f32_bf16<<<dim3((unsigned)blocks), dim3(256), 0, stream>>>(in, o, n);
    };
    auto cvtT = [&](const float* in, bf16_t* o, int R, int C) {
        int blocks = (R * C + 255) / 256;
        cvt_transpose_bf16<<<dim3(blocks), dim3(256), 0, stream>>>(in, o, R, C);
    };

    // ---- 1) convert activations; transpose-convert all weight (B) matrices ----
    cvt(x0, xb[0], nx[0]);
    cvt(x1, xb[1], nx[1]);
    cvt(x2, xb[2], nx[2]);
    for (int s = 0; s < 3; ++s) {
        int dim = 128 << s;
        cvtT(e_w1[s], wb1t[s], dim, dim);        // [dim,dim]  -> [dim,dim]^T
        cvtT(e_w2[s], wb2t[s], dim, 1024);       // [dim,1024] -> [1024,dim]
    }
    cvtT(conv1_w, conv1wbt, 1024, 1024);

    // ---- 2) embed MLPs (dominant GEMMs, WMMA bf16) ----
    for (int s = 0; s < 3; ++s) {
        int dim = 128 << s;
        // G1: y = relu(x @ w1 + b1)   [32768, dim]
        gemm_bf16_wmma<true, true><<<dim3(32768 / GEMM_BM, dim / GEMM_BN), dim3(256), 0, stream>>>(
            xb[s], wb1t[s], e_b1[s], (void*)ybf, 32768, dim, dim,
            /*rbShift=*/15, /*CS=*/0, /*ldc=*/dim);
        // G2: hs = y @ w2 + b2 -> scatter into [B,1536,1024] token-concat layout
        gemm_bf16_wmma<false, true><<<dim3(32768 / GEMM_BM, 1024 / GEMM_BN), dim3(256), 0, stream>>>(
            ybf, wb2t[s], e_b2[s], (void*)(hs_bf + (size_t)s * 512 * 1024), 32768, 1024, dim,
            /*rbShift=*/9, /*CS=*/1536L * 1024, /*ldc=*/1024);
    }

    // ---- 3) pool0 over 2048 tokens -> [B, C, 8] ----
    pool0_kernel<<<dim3(2, 64), dim3(256), 0, stream>>>(hs_bf, x3, pool0_w, pool0_b, pool_hs);

    // ---- 4) q/k channel means + attention adjacency ----
    qk_mean_kernel<<<dim3(9), dim3(256), 0, stream>>>(q_w, q_b, k_w, k_b, qkm);
    q1k1_kernel<<<dim3(4), dim3(256), 0, stream>>>(pool_hs, qkm, q1k1);
    a1_kernel<<<dim3(64), dim3(64), 0, stream>>>(q1k1, adj1, alpha1, A1b);

    // ---- 5) conv1 as one batched WMMA GEMM: [512,1024] x [1024,1024] ----
    hsT_kernel<<<dim3(2048), dim3(256), 0, stream>>>(pool_hs, hsT);
    gemm_bf16_wmma<false, false><<<dim3(512 / GEMM_BM, 1024 / GEMM_BN), dim3(256), 0, stream>>>(
        hsT, conv1wbt, conv1_b, (void*)hs2T, 512, 1024, 1024,
        /*rbShift=*/15, /*CS=*/0, /*ldc=*/1024);

    // ---- 6) graph matmul, BN, pool1, classifier ----
    graph_kernel<<<dim3(4, 64), dim3(256), 0, stream>>>(hs2T, A1b, hs3T);
    bn_stats_kernel<<<dim3(4), dim3(256), 0, stream>>>(hs3T, bn_gamma, bn_beta, bnscale, bnshift);
    pool1_kernel<<<dim3(256), dim3(256), 0, stream>>>(hs3T, bnscale, bnshift, pool1_w, pool1_b, hs4);
    cls_kernel<<<dim3(64), dim3(256), 0, stream>>>(hs4, cls_w, cls_b, out);
}
